// VectorQuantizer2_62886911148460
// MI455X (gfx1250) — compile-verified, matrix operand order checked by reading, not run on hardware
//
#include <hip/hip_runtime.h>
#include <hip/hip_bf16.h>

typedef _Float16 v16h __attribute__((ext_vector_type(16)));
typedef float    v8f  __attribute__((ext_vector_type(8)));
typedef unsigned int u32x4 __attribute__((ext_vector_type(4)));
typedef int          i32x8 __attribute__((ext_vector_type(8)));
typedef int          i32x4 __attribute__((ext_vector_type(4)));

// ---------------------------------------------------------------------------
// Codebook prep: pack emb (4096x32 f32) into f16 WMMA B-layout tiles + e^2.
// B tile (32x16, K x N): lane = n%16, g = lane/16.
// VGPR v (0..7) holds K = (v<4 ? 2v : 2v+8) + 8g  and K+1 (lo/hi halves).
// ---------------------------------------------------------------------------
__global__ __launch_bounds__(32) void prep_emb_kernel(const float* __restrict__ emb,
                                                      _Float16* __restrict__ embh,
                                                      float* __restrict__ e2) {
  int tile = blockIdx.x;          // 256 tiles of 16 codes
  int lane = threadIdx.x;         // wave32
  int n = tile * 16 + (lane & 15);
  int g = lane >> 4;
  const float* row = emb + (size_t)n * 32;
  float psum = 0.f;
  v16h b;
#pragma unroll
  for (int v = 0; v < 8; ++v) {
    int k0 = ((v < 4) ? 2 * v : 2 * v + 8) + 8 * g;
    float f0 = row[k0], f1 = row[k0 + 1];
    b[2 * v] = (_Float16)f0;
    b[2 * v + 1] = (_Float16)f1;
    psum += f0 * f0 + f1 * f1;
  }
  psum += __shfl_xor(psum, 16, 32);   // join the two K-halves (exact fp32 e^2)
  *(v16h*)(embh + ((size_t)(tile * 32 + lane)) * 16) = b;
  if (lane < 16) e2[n] = psum;
}

// ---------------------------------------------------------------------------
// Area downsample f_rest [32,32,32,32] -> token-major z [N,32], token=(b,y,x)
// ---------------------------------------------------------------------------
__global__ void downsample_kernel(const float* __restrict__ frest,
                                  float* __restrict__ rz, int pn) {
  int i = blockIdx.x * blockDim.x + threadIdx.x;
  int N = 32 * pn * pn;
  if (i >= N * 32) return;
  int c = i & 31;
  int token = i >> 5;
  int x = token % pn;
  int y = (token / pn) % pn;
  int b = token / (pn * pn);
  int s = 32 / pn;
  const float* base = frest + (((size_t)b * 32 + c) * 32 + y * s) * 32 + x * s;
  float acc = 0.f;
  for (int dy = 0; dy < s; ++dy)
    for (int dx = 0; dx < s; ++dx) acc += base[dy * 32 + dx];
  rz[i] = acc / (float)(s * s);
}

// ---------------------------------------------------------------------------
// WMMA nearest-codebook search with TDM-staged codebook in LDS.
// Block = 4 waves; each wave owns TWO 16-token tiles (2 accumulators per B).
// Codebook (256 tiles packed f16 B-layout) streamed in 4 chunks of 64 tiles
// (64KB) via tensor_load_to_lds, double-buffered; each wave DMAs its quarter.
// ---------------------------------------------------------------------------
#define CB_TILES 256
#define CHUNK    64
#define NCHUNK   (CB_TILES / CHUNK)

__device__ __forceinline__ void tdm_issue_chunk(_Float16* ldsdst,
                                                const _Float16* gsrc) {
  // 16 codebook tiles = 16*32*16 halves = 16384 B = 2048 elements of 8B.
  const unsigned E = 2048;
  unsigned ldsaddr = (unsigned)(uintptr_t)ldsdst;      // low 32 = LDS offset
  unsigned long long ga = (unsigned long long)(uintptr_t)gsrc;
  u32x4 g0;
  g0[0] = 1u;                                          // count=1, user mode
  g0[1] = ldsaddr;                                     // lds_addr
  g0[2] = (unsigned)(ga & 0xFFFFFFFFu);                // global_addr[31:0]
  g0[3] = (unsigned)((ga >> 32) & 0x1FFFFFFu) | (2u << 30);  // addr[56:32]|type=2
  i32x8 g1;
  g1[0] = (int)(3u << 16);      // workgroup_mask=0, data_size=3 (8B)
  g1[1] = (int)(E << 16);       // tensor_dim0[15:0]
  g1[2] = 0;                    // tensor_dim0[31:16]=0, tensor_dim1[15:0]=0
  g1[3] = (int)(E << 16);       // tensor_dim1[31:16]=0, tile_dim0=E
  g1[4] = 0;                    // tile_dim1=0, tile_dim2=0 (1-D copy)
  g1[5] = (int)E;               // tensor_dim0_stride[31:0]
  g1[6] = 0;
  g1[7] = 0;
  i32x4 g2 = {0, 0, 0, 0}, g3 = {0, 0, 0, 0};
  i32x8 g4 = {0, 0, 0, 0, 0, 0, 0, 0};
  __builtin_amdgcn_tensor_load_to_lds(g0, g1, g2, g3, g4, 0);
}

__global__ __launch_bounds__(128) void quantize_kernel(const float* __restrict__ rz,
                                                       const _Float16* __restrict__ embh,
                                                       const float* __restrict__ e2,
                                                       int* __restrict__ idx,
                                                       float* __restrict__ hits,
                                                       int ntiles) {
  __shared__ _Float16 sB[2][CHUNK * 32 * 16];   // 2 x 64KB double buffer

  int lane = threadIdx.x & 31;
  int wv = __builtin_amdgcn_readfirstlane(threadIdx.x >> 5);  // wave id 0..3
  int m = lane & 15;   // A: row, C/D: column n%16
  int g = lane >> 4;

  int pair = blockIdx.x * 4 + wv;
  int tile0 = pair * 2, tile1 = pair * 2 + 1;
  bool act0 = tile0 < ntiles, act1 = tile1 < ntiles;
  int tok0 = (act0 ? tile0 : 0) * 16 + m;
  int tok1 = (act1 ? tile1 : 0) * 16 + m;

  const float* r0 = rz + (size_t)tok0 * 32;
  const float* r1 = rz + (size_t)tok1 * 32;
  v16h a0, a1;
#pragma unroll
  for (int v = 0; v < 8; ++v) {
    int k0 = ((v < 4) ? 2 * v : 2 * v + 8) + 8 * g;
    a0[2 * v] = (_Float16)r0[k0];
    a0[2 * v + 1] = (_Float16)r0[k0 + 1];
    a1[2 * v] = (_Float16)r1[k0];
    a1[2 * v + 1] = (_Float16)r1[k0 + 1];
  }

  float best0[8], best1[8];
  int bi0[8], bi1[8];
#pragma unroll
  for (int r = 0; r < 8; ++r) {
    best0[r] = 3.4e38f; best1[r] = 3.4e38f;
    bi0[r] = 0; bi1[r] = 0;
  }

  // Each wave DMAs its quarter (16 tiles) of chunk 0 into buffer 0.
  tdm_issue_chunk(&sB[0][(wv * 16) * 32 * 16],
                  embh + ((size_t)(wv * 16)) * 32 * 16);

  for (int c = 0; c < NCHUNK; ++c) {
    __syncthreads();   // all waves done READING buffer (c+1)&1 from iter c-1
    if (c + 1 < NCHUNK) {
      tdm_issue_chunk(&sB[(c + 1) & 1][(wv * 16) * 32 * 16],
                      embh + ((size_t)((c + 1) * CHUNK + wv * 16)) * 32 * 16);
      __builtin_amdgcn_s_wait_tensorcnt(1);   // chunk c landed (in-order)
    } else {
      __builtin_amdgcn_s_wait_tensorcnt(0);
    }
    __syncthreads();   // chunk c visible to all waves

    const _Float16* bs = &sB[c & 1][0];
#pragma unroll 2
    for (int tt = 0; tt < CHUNK; ++tt) {
      v16h b = *(const v16h*)(bs + ((size_t)(tt * 32 + lane)) * 16);
      int nt = c * CHUNK + tt;
      float e2n = e2[nt * 16 + m];
      v8f c0 = {}, c1 = {};
      c0 = __builtin_amdgcn_wmma_f32_16x16x32_f16(false, a0, false, b, (short)0,
                                                  c0, false, false);
      c1 = __builtin_amdgcn_wmma_f32_16x16x32_f16(false, a1, false, b, (short)0,
                                                  c1, false, false);
      int nglob = nt * 16 + m;
#pragma unroll
      for (int r = 0; r < 8; ++r) {
        float s0 = e2n - 2.0f * c0[r];
        float s1 = e2n - 2.0f * c1[r];
        if (s0 < best0[r]) { best0[r] = s0; bi0[r] = nglob; }
        if (s1 < best1[r]) { best1[r] = s1; bi1[r] = nglob; }
      }
    }
  }

  // Cross-lane argmin over the 16 columns held by each half-wave, then write.
#pragma unroll
  for (int r = 0; r < 8; ++r) {
    float s0 = best0[r], s1 = best1[r];
    int i0 = bi0[r], i1 = bi1[r];
#pragma unroll
    for (int mask = 1; mask < 16; mask <<= 1) {
      float so0 = __shfl_xor(s0, mask, 32);
      int io0 = __shfl_xor(i0, mask, 32);
      if (so0 < s0 || (so0 == s0 && io0 < i0)) { s0 = so0; i0 = io0; }
      float so1 = __shfl_xor(s1, mask, 32);
      int io1 = __shfl_xor(i1, mask, 32);
      if (so1 < s1 || (so1 == s1 && io1 < i1)) { s1 = so1; i1 = io1; }
    }
    if ((lane & 15) == 0) {
      int row = r + 8 * g;   // D row held by VGPR r in this half-wave
      if (act0) {
        idx[tile0 * 16 + row] = i0;
        atomicAdd(&hits[i0], 1.0f);
      }
      if (act1) {
        idx[tile1 * 16 + row] = i1;
        atomicAdd(&hits[i1], 1.0f);
      }
    }
  }
}

// ---------------------------------------------------------------------------
// h = emb[idx] gathered into [B,C,pn,pn]
// ---------------------------------------------------------------------------
__global__ void gather_kernel(const float* __restrict__ emb,
                              const int* __restrict__ idx,
                              float* __restrict__ hout, int pn) {
  int i = blockIdx.x * blockDim.x + threadIdx.x;
  int total = 32 * 32 * pn * pn;
  if (i >= total) return;
  int x = i % pn;
  int t = i / pn;
  int y = t % pn;
  t /= pn;
  int c = t % 32;
  int b = t / 32;
  int token = (b * pn + y) * pn + x;
  hout[i] = emb[(size_t)idx[token] * 32 + c];
}

// ---------------------------------------------------------------------------
// Bicubic upsample (a=-0.75, half-pixel, edge clamp) pn -> 32
// ---------------------------------------------------------------------------
__device__ inline void cubw(float t, float w[4]) {
  const float A = -0.75f;
  float d;
  d = t + 1.0f; w[0] = ((A * d - 5.0f * A) * d + 8.0f * A) * d - 4.0f * A;
  d = t;        w[1] = ((A + 2.0f) * d - (A + 3.0f)) * d * d + 1.0f;
  d = 1.0f - t; w[2] = ((A + 2.0f) * d - (A + 3.0f)) * d * d + 1.0f;
  d = 2.0f - t; w[3] = ((A * d - 5.0f * A) * d + 8.0f * A) * d - 4.0f * A;
}

__global__ void upsample_kernel(const float* __restrict__ hsmall,
                                float* __restrict__ hup, int pn) {
  int i = blockIdx.x * blockDim.x + threadIdx.x;  // [B][C][32][32]
  if (i >= 32 * 32 * 32 * 32) return;
  int X = i & 31;
  int Y = (i >> 5) & 31;
  int bc = i >> 10;
  float scale = (float)pn / 32.0f;
  float fy = ((float)Y + 0.5f) * scale - 0.5f;
  float fx = ((float)X + 0.5f) * scale - 0.5f;
  int y0 = (int)floorf(fy), x0 = (int)floorf(fx);
  float wy[4], wx[4];
  cubw(fy - (float)y0, wy);
  cubw(fx - (float)x0, wx);
  const float* hp = hsmall + (size_t)bc * pn * pn;
  float acc = 0.f;
#pragma unroll
  for (int jy = 0; jy < 4; ++jy) {
    int sy = min(max(y0 + jy - 1, 0), pn - 1);
    float ra = 0.f;
#pragma unroll
    for (int jx = 0; jx < 4; ++jx) {
      int sx = min(max(x0 + jx - 1, 0), pn - 1);
      ra += wx[jx] * hp[sy * pn + sx];
    }
    acc += wy[jy] * ra;
  }
  hup[i] = acc;
}

// ---------------------------------------------------------------------------
// Phi + residual update: f_rest -= 0.5*h + 0.5*(conv3x3(h)+bias)
// One block per (b, y-row); 3 rows x 32ch x 32px staged in LDS.
// ---------------------------------------------------------------------------
__global__ __launch_bounds__(256) void phi_sub_kernel(const float* __restrict__ hup,
                                                      const float* __restrict__ w,
                                                      const float* __restrict__ bias,
                                                      float* __restrict__ frest) {
  __shared__ float s[3][32][32];
  int b = blockIdx.x >> 5;
  int y = blockIdx.x & 31;
  const float* hb = hup + (size_t)b * 32 * 32 * 32;
  for (int i = threadIdx.x; i < 3 * 32 * 32; i += 256) {
    int x = i & 31;
    int ci = (i >> 5) & 31;
    int ry = i >> 10;
    int gy = y + ry - 1;
    s[ry][ci][x] = (gy >= 0 && gy < 32) ? hb[((size_t)ci * 32 + gy) * 32 + x] : 0.f;
  }
  __syncthreads();
  int x = threadIdx.x & 31;
  int c0 = threadIdx.x >> 5;  // 0..7
  for (int cg = 0; cg < 4; ++cg) {
    int c = cg * 8 + c0;
    const float* wc = w + (size_t)c * 32 * 9;
    float acc = bias[c];
    for (int ci = 0; ci < 32; ++ci) {
      const float* wk = wc + ci * 9;
#pragma unroll
      for (int dy = 0; dy < 3; ++dy)
#pragma unroll
        for (int dx = 0; dx < 3; ++dx) {
          int gx = x + dx - 1;
          if (gx >= 0 && gx < 32) acc += wk[dy * 3 + dx] * s[dy][ci][gx];
        }
    }
    float out = 0.5f * s[1][c][x] + 0.5f * acc;
    frest[(((size_t)b * 32 + c) * 32 + y) * 32 + x] -= out;
  }
}

// ---------------------------------------------------------------------------
// init / loss / perplexity
// ---------------------------------------------------------------------------
__global__ void init_kernel(float* __restrict__ out, float* __restrict__ hits) {
  int i = blockIdx.x * blockDim.x + threadIdx.x;
  if (i < 1048578) out[i] = 0.f;  // f_hat (zeros) + loss + perplexity slots
  if (i < 4096) hits[i] = 0.f;
}

__global__ __launch_bounds__(256) void loss_kernel(const float* __restrict__ f,
                                                   float* __restrict__ lossout) {
  __shared__ float red[256];
  int i = blockIdx.x * 256 + threadIdx.x;  // exactly 4096 blocks
  float v = f[i];
  red[threadIdx.x] = v * v;
  __syncthreads();
  for (int st = 128; st > 0; st >>= 1) {
    if (threadIdx.x < st) red[threadIdx.x] += red[threadIdx.x + st];
    __syncthreads();
  }
  // 6 scales * (0.25 + 1.0) * mean(f^2), f_hat is identically zero
  if (threadIdx.x == 0) atomicAdd(lossout, red[0] * (7.5f / 1048576.0f));
}

__global__ __launch_bounds__(256) void perplexity_kernel(const float* __restrict__ hits,
                                                         float* __restrict__ out) {
  __shared__ float red[256];
  float t = 0.f;
  for (int i = threadIdx.x; i < 4096; i += 256) t += hits[i];
  red[threadIdx.x] = t;
  __syncthreads();
  for (int st = 128; st > 0; st >>= 1) {
    if (threadIdx.x < st) red[threadIdx.x] += red[threadIdx.x + st];
    __syncthreads();
  }
  float total = fmaxf(red[0], 1.0f);
  __syncthreads();
  float h = 0.f;
  for (int i = threadIdx.x; i < 4096; i += 256) {
    float p = hits[i] / total;
    h += p * logf(p + 1e-10f);
  }
  red[threadIdx.x] = h;
  __syncthreads();
  for (int st = 128; st > 0; st >>= 1) {
    if (threadIdx.x < st) red[threadIdx.x] += red[threadIdx.x + st];
    __syncthreads();
  }
  if (threadIdx.x == 0) out[0] = expf(-red[0]);
}

// ---------------------------------------------------------------------------
extern "C" void kernel_launch(void* const* d_in, const int* in_sizes, int n_in,
                              void* d_out, int out_size, void* d_ws, size_t ws_size,
                              hipStream_t stream) {
  const float* f     = (const float*)d_in[0];  // [32,32,32,32]
  const float* emb   = (const float*)d_in[1];  // [4096,32]
  const float* phi_w = (const float*)d_in[2];  // [4,32,32,3,3]
  const float* phi_b = (const float*)d_in[3];  // [4,32]
  float* out = (float*)d_out;                  // f_hat(1048576) + loss + perplexity

  char* ws = (char*)d_ws;                      // ~13.4 MB, all L2-resident
  float*    frest  = (float*)(ws + 0);                         // 4 MB
  float*    rz     = (float*)(ws + (1u << 22));                // 4 MB
  _Float16* embh   = (_Float16*)(ws + (2u << 22));             // 256 KB
  float*    e2     = (float*)(ws + (2u << 22) + 262144);       // 16 KB
  int*      idx    = (int*)(ws + (2u << 22) + 278528);         // 128 KB
  float*    hits   = (float*)(ws + (2u << 22) + 409600);       // 16 KB
  float*    hsmall = (float*)(ws + (2u << 22) + 425984);       // 1 MB
  float*    hup    = (float*)(ws + (2u << 22) + 1474560);      // 4 MB

  (void)hipMemcpyAsync(frest, f, (size_t)1048576 * 4, hipMemcpyDeviceToDevice, stream);
  init_kernel<<<(1048578 + 255) / 256, 256, 0, stream>>>(out, hits);
  loss_kernel<<<4096, 256, 0, stream>>>(f, out + 1048576);
  prep_emb_kernel<<<256, 32, 0, stream>>>(emb, embh, e2);

  const int PN[6]  = {1, 2, 4, 8, 16, 32};
  const int SEL[6] = {0, 0, 1, 2, 3, 3};  // PHI_SEL
  for (int si = 0; si < 6; ++si) {
    int pn = PN[si];
    int N = 32 * pn * pn;
    downsample_kernel<<<(N * 32 + 255) / 256, 256, 0, stream>>>(frest, rz, pn);
    int ntiles = N / 16;
    int pairs = ntiles / 2;
    quantize_kernel<<<(pairs + 3) / 4, 128, 0, stream>>>(rz, embh, e2, idx, hits, ntiles);
    float* hdst = (si == 5) ? hup : hsmall;
    gather_kernel<<<(N * 32 + 255) / 256, 256, 0, stream>>>(emb, idx, hdst, pn);
    if (si != 5)
      upsample_kernel<<<(1048576 + 255) / 256, 256, 0, stream>>>(hsmall, hup, pn);
    phi_sub_kernel<<<1024, 256, 0, stream>>>(
        hup, phi_w + (size_t)SEL[si] * 32 * 32 * 9, phi_b + SEL[si] * 32, frest);
  }
  perplexity_kernel<<<1, 256, 0, stream>>>(hits, out + 1048577);
}
